// CrossMultiHeadAttention_24515673325888
// MI455X (gfx1250) — matrix operand order repaired and checked by reading, not compiled, not run on hardware
//
#include <hip/hip_runtime.h>
#include <hip/hip_bf16.h>

// CrossMultiHeadAttention for MI455X (gfx1250), bf16 WMMA pipeline.
// Inputs (setup_inputs order): x[2,2048,1024] f32, K[2,16,2048,64] f32,
// V[2,16,2048,64] f32, Wq[1024,1024] f32, Wo[1024,1024] f32,
// kv_pad_mask[2,1,1,2048] bool(u8), q_pad_mask[2,1,2048,1] bool(u8).
// Output: [2,2048,1024] f32.

#define NQ    2048
#define NKV   2048
#define NH    16
#define DK    64
#define DM    1024
#define BATCH 2

typedef __attribute__((ext_vector_type(16))) __bf16 bf16x16;
typedef __attribute__((ext_vector_type(8)))  float  f32x8;

union Frag32B { uint4 u[2]; bf16x16 v; };

__device__ __forceinline__ f32x8 wmma_bf16(bf16x16 a, bf16x16 b, f32x8 c) {
  // D = A(16x32 bf16) * B(32x16 bf16) + C(16x16 f32)
  return __builtin_amdgcn_wmma_f32_16x16x32_bf16(false, a, false, b,
                                                 (short)0, c, false, false);
}

// A-matrix 16x32 bf16 fragment (ISA 7.12.2): lane m<16 -> row m,
// VGPR0..3 = K {ba..ba+7}, VGPR4..7 = K {ba+16..ba+23}, ba = (lane<16)?0:8.
__device__ __forceinline__ bf16x16 load_frag_a(const __bf16* base, int ldm, int lane) {
  const __bf16* p = base + (long)(lane & 15) * ldm + ((lane & 16) ? 8 : 0);
  Frag32B f;
  f.u[0] = *(const uint4*)(p);
  f.u[1] = *(const uint4*)(p + 16);
  return f.v;
}

// B-matrix 32x16 bf16 fragment: lane n<16 -> column n,
// VGPR0..3 = K {kb..kb+7}, VGPR4..7 = K {kb+16..kb+23}, kb = (lane<16)?0:8.
// `base` points at (k=0, n=0); columns are contiguous along k with stride col_stride.
__device__ __forceinline__ bf16x16 load_frag_b(const __bf16* base, int col_stride, int lane) {
  const __bf16* p = base + (long)(lane & 15) * col_stride + ((lane & 16) ? 8 : 0);
  Frag32B f;
  f.u[0] = *(const uint4*)(p);
  f.u[1] = *(const uint4*)(p + 16);
  return f.v;
}

// Cross-lane reductions within each 16-lane half (rows of the C fragment
// live across lanes 0..15 / 16..31). ds_swizzle xor patterns stay in-half.
__device__ __forceinline__ float red_max16(float v) {
  v = fmaxf(v, __int_as_float(__builtin_amdgcn_ds_swizzle(__float_as_int(v), 0x041f)));
  v = fmaxf(v, __int_as_float(__builtin_amdgcn_ds_swizzle(__float_as_int(v), 0x081f)));
  v = fmaxf(v, __int_as_float(__builtin_amdgcn_ds_swizzle(__float_as_int(v), 0x101f)));
  v = fmaxf(v, __int_as_float(__builtin_amdgcn_ds_swizzle(__float_as_int(v), 0x201f)));
  return v;
}
__device__ __forceinline__ float red_sum16(float v) {
  v += __int_as_float(__builtin_amdgcn_ds_swizzle(__float_as_int(v), 0x041f));
  v += __int_as_float(__builtin_amdgcn_ds_swizzle(__float_as_int(v), 0x081f));
  v += __int_as_float(__builtin_amdgcn_ds_swizzle(__float_as_int(v), 0x101f));
  v += __int_as_float(__builtin_amdgcn_ds_swizzle(__float_as_int(v), 0x201f));
  return v;
}

// ---------------- elementwise f32 -> bf16 ----------------
__global__ void __launch_bounds__(256) k_convert(const float* __restrict__ in,
                                                 __bf16* __restrict__ out, long n) {
  long i = (long)blockIdx.x * blockDim.x + threadIdx.x;
  long stride = (long)gridDim.x * blockDim.x;
  for (; i < n; i += stride) out[i] = (__bf16)in[i];
}

// -------- batched transpose + convert: in[b][R][C] f32 -> out[b][C][R] bf16 --------
__global__ void __launch_bounds__(256) k_tconv(const float* __restrict__ in,
                                               __bf16* __restrict__ out, int R, int C) {
  __shared__ float t[32][33];
  const int tx = threadIdx.x, ty = threadIdx.y;
  const long boff = (long)blockIdx.z * R * C;
  const int r0 = blockIdx.y * 32, c0 = blockIdx.x * 32;
#pragma unroll
  for (int i = 0; i < 4; ++i)
    t[ty + i * 8][tx] = in[boff + (long)(r0 + ty + i * 8) * C + c0 + tx];
  __syncthreads();
#pragma unroll
  for (int i = 0; i < 4; ++i)
    out[boff + (long)(c0 + ty + i * 8) * R + r0 + tx] = (__bf16)t[tx][ty + i * 8];
}

// ---------------- GEMM: C[M,N] = scale * A[M,K] * Bt[N,K]^T ----------------
// 256 threads = 8 waves; wave computes 32x64; block tile 256x64.
__global__ void __launch_bounds__(256) k_gemm(const __bf16* __restrict__ A,
                                              const __bf16* __restrict__ Bt,
                                              int M, int N, int K, float scale,
                                              float* __restrict__ Cf,
                                              __bf16* __restrict__ Cb) {
  const int lane = threadIdx.x & 31;
  const int wave = threadIdx.x >> 5;
  const int n0 = blockIdx.x * 64;
  const int m0 = blockIdx.y * 256 + wave * 32;

  f32x8 acc[2][4];
#pragma unroll
  for (int mt = 0; mt < 2; ++mt)
#pragma unroll
    for (int nt = 0; nt < 4; ++nt)
#pragma unroll
      for (int r = 0; r < 8; ++r) acc[mt][nt][r] = 0.f;

  for (int k0 = 0; k0 < K; k0 += 32) {
    bf16x16 a0 = load_frag_a(A + (long)m0 * K + k0, K, lane);
    bf16x16 a1 = load_frag_a(A + (long)(m0 + 16) * K + k0, K, lane);
#pragma unroll
    for (int nt = 0; nt < 4; ++nt) {
      bf16x16 bfr = load_frag_b(Bt + (long)(n0 + nt * 16) * K + k0, K, lane);
      acc[0][nt] = wmma_bf16(a0, bfr, acc[0][nt]);
      acc[1][nt] = wmma_bf16(a1, bfr, acc[1][nt]);
    }
  }

#pragma unroll
  for (int mt = 0; mt < 2; ++mt)
#pragma unroll
    for (int nt = 0; nt < 4; ++nt)
#pragma unroll
      for (int r = 0; r < 8; ++r) {
        int row = m0 + mt * 16 + r + ((lane & 16) ? 8 : 0);
        int col = n0 + nt * 16 + (lane & 15);
        float v = acc[mt][nt][r] * scale;
        if (Cf) Cf[(long)row * N + col] = v;
        else    Cb[(long)row * N + col] = (__bf16)v;
      }
}

// ---------------- Flash attention ----------------
// grid = (NQ/128, NH, BATCH), block = 256 (8 waves). Wave owns 16 q rows.
// Qb: [B*NQ, DM] bf16 (pre-scaled by 1/sqrt(dk)); Kb: [B*H*NKV, DK] bf16;
// Vt: [B*H*DK, NKV] bf16 (transposed); attn out: [B*NQ, DM] bf16.
__global__ void __launch_bounds__(256) k_attn(const __bf16* __restrict__ Qb,
                                              const __bf16* __restrict__ Kb,
                                              const __bf16* __restrict__ Vt,
                                              const unsigned char* __restrict__ kvm,
                                              const unsigned char* __restrict__ qpm,
                                              __bf16* __restrict__ attn) {
  const int lane = threadIdx.x & 31;
  const int wave = threadIdx.x >> 5;
  const int h = blockIdx.y;
  const int b = blockIdx.z;
  const int q0 = blockIdx.x * 128 + wave * 16;   // row within batch
  const long qrow0 = (long)b * NQ + q0;          // global row
  const int bh = b * NH + h;

  __shared__ __align__(16) __bf16 lds_p[8][16 * 32];
  __bf16* myp = &lds_p[wave][0];

  // Q fragments held for the whole kv loop (d 0..31, 32..63)
  const bf16x16 qa0 = load_frag_a(Qb + qrow0 * DM + h * DK + 0,  DM, lane);
  const bf16x16 qa1 = load_frag_a(Qb + qrow0 * DM + h * DK + 32, DM, lane);

  float qmask[8];
#pragma unroll
  for (int r = 0; r < 8; ++r) {
    int row = r + ((lane & 16) ? 8 : 0);
    qmask[r] = qpm[(long)b * NQ + q0 + row] ? 1.f : 0.f;
  }

  f32x8 acc[4];
#pragma unroll
  for (int nt = 0; nt < 4; ++nt)
#pragma unroll
    for (int r = 0; r < 8; ++r) acc[nt][r] = 0.f;
  float mrow[8], lrow[8];
#pragma unroll
  for (int r = 0; r < 8; ++r) { mrow[r] = -1e30f; lrow[r] = 0.f; }

  for (int kv0 = 0; kv0 < NKV; kv0 += 32) {
    float s[2][8];
#pragma unroll
    for (int t = 0; t < 2; ++t) {
      const int kvb = kv0 + t * 16;
      const __bf16* kbase = Kb + ((long)bh * NKV + kvb) * DK;
      bf16x16 kb0 = load_frag_b(kbase + 0,  DK, lane);
      bf16x16 kb1 = load_frag_b(kbase + 32, DK, lane);
      f32x8 c;
#pragma unroll
      for (int r = 0; r < 8; ++r) c[r] = 0.f;
      c = wmma_bf16(qa0, kb0, c);
      c = wmma_bf16(qa1, kb1, c);
      float kvok = kvm[(long)b * NKV + kvb + (lane & 15)] ? 1.f : 0.f;
#pragma unroll
      for (int r = 0; r < 8; ++r)
        s[t][r] = (kvok * qmask[r] != 0.f) ? c[r] : -1e9f;
    }

    // online softmax (rows live across lanes within each 16-lane half)
#pragma unroll
    for (int r = 0; r < 8; ++r) {
      float mx = red_max16(fmaxf(s[0][r], s[1][r]));
      float mn = fmaxf(mrow[r], mx);
      float alpha = __expf(mrow[r] - mn);
      float p0 = __expf(s[0][r] - mn);
      float p1 = __expf(s[1][r] - mn);
      s[0][r] = p0; s[1][r] = p1;
      float rs = red_sum16(p0 + p1);
      lrow[r] = lrow[r] * alpha + rs;
      mrow[r] = mn;
#pragma unroll
      for (int nt = 0; nt < 4; ++nt) acc[nt][r] *= alpha;
    }

    // C-layout P -> LDS (row-major 16x32) -> A-layout fragment
#pragma unroll
    for (int t = 0; t < 2; ++t)
#pragma unroll
      for (int r = 0; r < 8; ++r) {
        int row = r + ((lane & 16) ? 8 : 0);
        myp[row * 32 + t * 16 + (lane & 15)] = (__bf16)s[t][r];
      }
    asm volatile("s_wait_dscnt 0" ::: "memory");
    bf16x16 pa = load_frag_a(myp, 32, lane);

#pragma unroll
    for (int nt = 0; nt < 4; ++nt) {
      const __bf16* vbase = Vt + ((long)bh * DK + nt * 16) * NKV + kv0;
      bf16x16 vb = load_frag_b(vbase, NKV, lane);
      acc[nt] = wmma_bf16(pa, vb, acc[nt]);
    }
  }

  // normalize and store attn tile as bf16 [B*NQ, DM] at column h*64
#pragma unroll
  for (int r = 0; r < 8; ++r) {
    float inv = 1.f / lrow[r];
    int row = r + ((lane & 16) ? 8 : 0);
    long obase = (qrow0 + row) * DM + h * DK;
#pragma unroll
    for (int nt = 0; nt < 4; ++nt)
      attn[obase + nt * 16 + (lane & 15)] = (__bf16)(acc[nt][r] * inv);
  }
}

extern "C" void kernel_launch(void* const* d_in, const int* in_sizes, int n_in,
                              void* d_out, int out_size, void* d_ws, size_t ws_size,
                              hipStream_t stream) {
  const float* x  = (const float*)d_in[0];
  const float* Ki = (const float*)d_in[1];
  const float* Vi = (const float*)d_in[2];
  const float* Wq = (const float*)d_in[3];
  const float* Wo = (const float*)d_in[4];
  const unsigned char* kvm = (const unsigned char*)d_in[5];  // bool, 1B each
  const unsigned char* qpm = (const unsigned char*)d_in[6];
  float* out = (float*)d_out;

  char* ws = (char*)d_ws;
  __bf16* x_bf  = (__bf16*)(ws);                      // 8 MB (reused as attn)
  __bf16* K_bf  = (__bf16*)(ws + (8u  << 20));        // 8 MB
  __bf16* Vt_bf = (__bf16*)(ws + (16u << 20));        // 8 MB
  __bf16* WqT   = (__bf16*)(ws + (24u << 20));        // 2 MB
  __bf16* WoT   = (__bf16*)(ws + (26u << 20));        // 2 MB
  __bf16* Q_bf  = (__bf16*)(ws + (28u << 20));        // 8 MB

  const long nX = (long)BATCH * NQ * DM;              // 4M
  const long nK = (long)BATCH * NH * NKV * DK;        // 4M

  // 1) convert x, K to bf16 (layouts unchanged)
  k_convert<<<2048, 256, 0, stream>>>(x,  x_bf, nX);
  k_convert<<<2048, 256, 0, stream>>>(Ki, K_bf, nK);

  // 2) transpose+convert: Wq,Wo -> [N,K]; V -> [d,kv] per (b,h)
  k_tconv<<<dim3(32, 32, 1),          dim3(32, 8), 0, stream>>>(Wq, WqT,  DM,  DM);
  k_tconv<<<dim3(32, 32, 1),          dim3(32, 8), 0, stream>>>(Wo, WoT,  DM,  DM);
  k_tconv<<<dim3(2, 64, BATCH * NH),  dim3(32, 8), 0, stream>>>(Vi, Vt_bf, NKV, DK);

  // 3) Q = (1/sqrt(dk)) * x @ Wq  -> bf16
  k_gemm<<<dim3(DM / 64, (BATCH * NQ) / 256), 256, 0, stream>>>(
      x_bf, WqT, BATCH * NQ, DM, DM, 0.125f, nullptr, Q_bf);

  // 4) flash attention -> attn bf16 (reuses x_bf buffer)
  k_attn<<<dim3(NQ / 128, NH, BATCH), 256, 0, stream>>>(
      Q_bf, K_bf, Vt_bf, kvm, qpm, x_bf);

  // 5) out = attn @ Wo -> f32
  k_gemm<<<dim3(DM / 64, (BATCH * NQ) / 256), 256, 0, stream>>>(
      x_bf, WoT, BATCH * NQ, DM, DM, 1.0f, out, nullptr);
}
